// DAGGNNRecommender_16947940950532
// MI455X (gfx1250) — matrix operand order repaired and checked by reading, not compiled, not run on hardware
//
#include <hip/hip_runtime.h>
#include <hip/hip_bf16.h>

// ---------------------------------------------------------------------------
// DAG-GNN recommender, CDNA5 (gfx1250, wave32).
//  * aggregate-then-transform:  segsum(h[src] @ W) == segsum(h[src]) @ W
//    (cuts aggregation matmuls from E*H*H to N*H*H FLOPs, 16x)
//  * per-layer fused GEMM:      out = [h | aggP | aggC] (N x 768) @ stacked W
//  * CSR rebuilt per call (stateless) -> atomic-free gather aggregation
//  * GEMMs on v_wmma_f32_16x16x32_bf16; operands pre-staged in bf16 so the
//    k-loop is pure b128 loads + WMMA.  Working set (~70 MB) is L2-resident.
//  * aggregation: one wave32 per node, one b128 (8 bf16 cols) per lane/edge.
// ---------------------------------------------------------------------------

typedef __attribute__((ext_vector_type(16))) __bf16        v16bf;
typedef __attribute__((ext_vector_type(8)))  float         v8f;
typedef __attribute__((ext_vector_type(8)))  unsigned int  v8u;

#define HDIM 256
#define BIGLD 768   // bf16 activation buffer: [h | aggP | aggC]

__device__ __forceinline__ unsigned short f2bf(float x) {
    unsigned int u = __builtin_bit_cast(unsigned int, x);
    u = (u + 0x7FFFu + ((u >> 16) & 1u)) >> 16;             // RNE f32 -> bf16
    return (unsigned short)u;
}
__device__ __forceinline__ unsigned int pk2bf(float lo, float hi) {
    return ((unsigned int)f2bf(hi) << 16) | (unsigned int)f2bf(lo);
}
__device__ __forceinline__ float gelu_exact(float x) {
    return 0.5f * x * (1.0f + erff(x * 0.70710678118654752f));
}

// ---------------------------------------------------------------------------
// elementwise f32 -> bf16 cast (x -> xb), 4 elements per thread
// ---------------------------------------------------------------------------
__global__ void cast_kernel(const float4* __restrict__ src,
                            uint2* __restrict__ dst, long n4) {
    long i = (long)blockIdx.x * blockDim.x + threadIdx.x;
    if (i < n4) {
        float4 f = src[i];
        uint2 o;
        o.x = pk2bf(f.x, f.y);
        o.y = pk2bf(f.z, f.w);
        dst[i] = o;
    }
}

// ---------------------------------------------------------------------------
// Pack a K x Ncols f32 weight matrix into the wave32 WMMA B-fragment layout
// (B tile 32x16 KxN: lanes 0-15 hold K={0..7,16..23}, VGPR j -> 2 K; lanes 16-31
// hold K={8..15,24..31}).  Lane-major storage: dword (f,lane,j) at dst[(f*32+lane)*8+j]
// so a wave fetches one fragment with two b128 loads per lane.
// ktOff stacks multiple matrices along K into one packed buffer.
// ---------------------------------------------------------------------------
__global__ void pack_b_kernel(const float* __restrict__ w, int ldw,
                              int nRows, int nCols,
                              unsigned int* __restrict__ dst,
                              int ktOff, int KT) {
    int t = blockIdx.x * blockDim.x + threadIdx.x;
    int total = (nRows / 32) * (nCols / 16) * 8 * 32;
    if (t >= total) return;
    int j    = t & 7;
    int lane = (t >> 3) & 31;
    int fl   = t >> 8;
    int nKt  = nRows / 32;
    int ktl  = fl % nKt;
    int nt   = fl / nKt;
    int row  = ktl * 32 + ((lane & 16) ? 8 : 0) + 2 * j + ((j >= 4) ? 8 : 0);
    int col  = nt * 16 + (lane & 15);
    long f   = (long)nt * KT + ktOff + ktl;
    dst[((f * 32) + lane) * 8 + j] =
        pk2bf(w[(long)row * ldw + col], w[(long)(row + 1) * ldw + col]);
}

// ---------------------------------------------------------------------------
// GEMM: D[M x Ncols] = gelu?( A[M x K] @ B + bias ),  K = KT*32, A in bf16.
// One wave -> 16x64 strip (4 v8f accumulators); 8 waves/block -> 128x64.
// A-frag (16-bit A 16x32): lane<16 -> M=lane, K={0..7,16..23};
// lane>=16 -> M=lane-16, K={8..15,24..31}.  Two uint4 loads per k-tile.
// flags: bit0 = gelu, bit1 = write outF (f32), bit2 = write outB (bf16).
// ---------------------------------------------------------------------------
__global__ __launch_bounds__(256)
void wmma_gemm_kernel(const unsigned short* __restrict__ A, int lda,
                      const uint4* __restrict__ Bp, int KT,
                      const float* __restrict__ bias,
                      float* __restrict__ outF, int ldF,
                      unsigned short* __restrict__ outB, int ldB,
                      int M, int flags) {
    const int lane = threadIdx.x & 31;
    const int wv   = threadIdx.x >> 5;
    const int mTile = (blockIdx.y * 8 + wv) * 16;
    if (mTile >= M) return;                      // whole-wave exit: EXEC stays all-1
    const int nBase = blockIdx.x * 64;

    v8f acc[4] = {v8f{}, v8f{}, v8f{}, v8f{}};

    const unsigned short* __restrict__ arow =
        A + (long)(mTile + (lane & 15)) * lda + ((lane & 16) ? 8 : 0);

    for (int kt = 0; kt < KT; ++kt) {
        uint4 alo = *(const uint4*)(arow + kt * 32);        // K kb..kb+7
        uint4 ahi = *(const uint4*)(arow + kt * 32 + 16);   // K kb+16..kb+23
        v8u au = {alo.x, alo.y, alo.z, alo.w, ahi.x, ahi.y, ahi.z, ahi.w};
        v16bf afrag = __builtin_bit_cast(v16bf, au);

#pragma unroll
        for (int i = 0; i < 4; ++i) {
            const long f = (long)((nBase >> 4) + i) * KT + kt;
            const uint4* bsrc = Bp + (f * 32 + lane) * 2;
            uint4 b0 = bsrc[0];
            uint4 b1 = bsrc[1];
            v8u bu = {b0.x, b0.y, b0.z, b0.w, b1.x, b1.y, b1.z, b1.w};
            v16bf bfrag = __builtin_bit_cast(v16bf, bu);
            acc[i] = __builtin_amdgcn_wmma_f32_16x16x32_bf16(
                false, afrag, false, bfrag, (short)0, acc[i], false, false);
        }
    }

    // C/D layout: VGPR r: lane<16 -> M=r,N=lane ; lane>=16 -> M=8+r,N=lane-16.
    const int mOff = mTile + ((lane & 16) ? 8 : 0);
#pragma unroll
    for (int i = 0; i < 4; ++i) {
        const int col = nBase + i * 16 + (lane & 15);
        const float b = bias[col];
#pragma unroll
        for (int r = 0; r < 8; ++r) {
            float v = acc[i][r] + b;
            if (flags & 1) v = gelu_exact(v);
            if (flags & 2) outF[(long)(mOff + r) * ldF + col] = v;
            if (flags & 4) outB[(long)(mOff + r) * ldB + col] = f2bf(v);
        }
    }
}

// ---------------------------------------------------------------------------
// CSR construction (rebuilt every call -> stateless / capture-safe)
// ---------------------------------------------------------------------------
__global__ void count_kernel(const int* __restrict__ ei, int E,
                             int* __restrict__ cntD, int* __restrict__ cntS) {
    int e = blockIdx.x * blockDim.x + threadIdx.x;
    if (e >= E) return;
    atomicAdd(&cntD[ei[E + e]], 1);   // in-degree  (dst)
    atomicAdd(&cntS[ei[e]], 1);       // out-degree (src)
}

__global__ void scan_kernel(const int* __restrict__ cnt,
                            int* __restrict__ rowptr, int n) {
    __shared__ int sdata[1024];
    __shared__ int carry;
    if (threadIdx.x == 0) { carry = 0; rowptr[0] = 0; }
    __syncthreads();
    for (int base = 0; base < n; base += 1024) {
        int i = base + (int)threadIdx.x;
        sdata[threadIdx.x] = (i < n) ? cnt[i] : 0;
        __syncthreads();
        for (int off = 1; off < 1024; off <<= 1) {
            int t = (threadIdx.x >= (unsigned)off) ? sdata[threadIdx.x - off] : 0;
            __syncthreads();
            sdata[threadIdx.x] += t;
            __syncthreads();
        }
        int inc = sdata[threadIdx.x] + carry;
        if (i < n) rowptr[i + 1] = inc;
        __syncthreads();
        if (threadIdx.x == 1023) carry = inc;
        __syncthreads();
    }
}

__global__ void copy2_kernel(const int* __restrict__ a, int* __restrict__ ca,
                             const int* __restrict__ b, int* __restrict__ cb, int n) {
    int i = blockIdx.x * blockDim.x + threadIdx.x;
    if (i < n) { ca[i] = a[i]; cb[i] = b[i]; }
}

__global__ void scatter_kernel(const int* __restrict__ ei, int E,
                               int* __restrict__ curD, int* __restrict__ adjD,
                               int* __restrict__ curS, int* __restrict__ adjS) {
    int e = blockIdx.x * blockDim.x + threadIdx.x;
    if (e >= E) return;
    int s = ei[e], d = ei[E + e];
    adjD[atomicAdd(&curD[d], 1)] = s;   // parents of d
    adjS[atomicAdd(&curS[s], 1)] = d;   // children of s
}

// ---------------------------------------------------------------------------
// Gather-reduce aggregation with degree normalization (the hot L2 stream).
// One wave32 per node; lane owns 8 contiguous bf16 columns -> each edge is a
// single b128 load per lane (512 B per wave-instruction).  f32 accumulation;
// bf16 unpack is 1 shift + 1 mask per dword.  Outputs stored as packed uint4.
// bigb row: [h(256) | aggP(256) | aggC(256)].
// ---------------------------------------------------------------------------
__global__ __launch_bounds__(256)
void agg_kernel(unsigned short* __restrict__ bigb,
                const int* __restrict__ rpD, const int* __restrict__ adjD,
                const int* __restrict__ rpS, const int* __restrict__ adjS,
                int n) {
    const int  lane = threadIdx.x & 31;
    const long node = (long)blockIdx.x * 8 + (threadIdx.x >> 5);
    if (node >= n) return;
    const int cb = lane * 8;            // this lane's column base (0..248)

#pragma unroll
    for (int half = 0; half < 2; ++half) {
        const int* rp  = half ? rpS : rpD;
        const int* adj = half ? adjS : adjD;
        int e0 = rp[node], e1 = rp[node + 1];
        float acc[8];
#pragma unroll
        for (int i = 0; i < 8; ++i) acc[i] = 0.f;
        for (int e = e0; e < e1; ++e) {
            const uint4 u = *(const uint4*)(bigb + (long)adj[e] * BIGLD + cb);
            unsigned int w[4] = {u.x, u.y, u.z, u.w};
#pragma unroll
            for (int j = 0; j < 4; ++j) {
                acc[2 * j]     += __builtin_bit_cast(float, w[j] << 16);
                acc[2 * j + 1] += __builtin_bit_cast(float, w[j] & 0xFFFF0000u);
            }
        }
        float inv = 1.0f / (float)((e1 - e0) > 1 ? (e1 - e0) : 1);
        uint4 o;
        o.x = pk2bf(acc[0] * inv, acc[1] * inv);
        o.y = pk2bf(acc[2] * inv, acc[3] * inv);
        o.z = pk2bf(acc[4] * inv, acc[5] * inv);
        o.w = pk2bf(acc[6] * inv, acc[7] * inv);
        *(uint4*)(bigb + node * BIGLD + 256 + half * 256 + cb) = o;
    }
}

// ---------------------------------------------------------------------------
// h = LayerNorm(h + t) * g + b (f32), then write f32 h and bf16 mirror.
// One wave32 per row; lane owns 8 contiguous columns (vectorized loads/stores);
// shfl_xor reductions across the wave.
// ---------------------------------------------------------------------------
__global__ __launch_bounds__(256)
void ln_kernel(float* __restrict__ hbuf, const float* __restrict__ t,
               unsigned short* __restrict__ bigb,
               const float* __restrict__ g, const float* __restrict__ b, int n) {
    const int lane = threadIdx.x & 31;
    const long row = (long)blockIdx.x * 8 + (threadIdx.x >> 5);
    if (row >= n) return;
    const int cb = lane * 8;

    float4 h0 = *(const float4*)(hbuf + row * HDIM + cb);
    float4 h1 = *(const float4*)(hbuf + row * HDIM + cb + 4);
    float4 t0 = *(const float4*)(t + row * HDIM + cb);
    float4 t1 = *(const float4*)(t + row * HDIM + cb + 4);
    float v[8] = {h0.x + t0.x, h0.y + t0.y, h0.z + t0.z, h0.w + t0.w,
                  h1.x + t1.x, h1.y + t1.y, h1.z + t1.z, h1.w + t1.w};
    float sum = 0.f;
#pragma unroll
    for (int i = 0; i < 8; ++i) sum += v[i];
#pragma unroll
    for (int m = 16; m >= 1; m >>= 1) sum += __shfl_xor(sum, m, 32);
    float mean = sum * (1.0f / 256.0f);
    float var = 0.f;
#pragma unroll
    for (int i = 0; i < 8; ++i) { float d = v[i] - mean; var += d * d; }
#pragma unroll
    for (int m = 16; m >= 1; m >>= 1) var += __shfl_xor(var, m, 32);
    var *= (1.0f / 256.0f);
    float inv = rsqrtf(var + 1e-5f);

    float4 g0 = *(const float4*)(g + cb);
    float4 g1 = *(const float4*)(g + cb + 4);
    float4 b0 = *(const float4*)(b + cb);
    float4 b1 = *(const float4*)(b + cb + 4);
    float y[8];
    y[0] = (v[0] - mean) * inv * g0.x + b0.x;
    y[1] = (v[1] - mean) * inv * g0.y + b0.y;
    y[2] = (v[2] - mean) * inv * g0.z + b0.z;
    y[3] = (v[3] - mean) * inv * g0.w + b0.w;
    y[4] = (v[4] - mean) * inv * g1.x + b1.x;
    y[5] = (v[5] - mean) * inv * g1.y + b1.y;
    y[6] = (v[6] - mean) * inv * g1.z + b1.z;
    y[7] = (v[7] - mean) * inv * g1.w + b1.w;

    *(float4*)(hbuf + row * HDIM + cb)     = make_float4(y[0], y[1], y[2], y[3]);
    *(float4*)(hbuf + row * HDIM + cb + 4) = make_float4(y[4], y[5], y[6], y[7]);
    uint4 o;
    o.x = pk2bf(y[0], y[1]);
    o.y = pk2bf(y[2], y[3]);
    o.z = pk2bf(y[4], y[5]);
    o.w = pk2bf(y[6], y[7]);
    *(uint4*)(bigb + row * BIGLD + cb) = o;
}

// ---------------------------------------------------------------------------
extern "C" void kernel_launch(void* const* d_in, const int* in_sizes, int n_in,
                              void* d_out, int out_size, void* d_ws, size_t ws_size,
                              hipStream_t stream) {
    (void)n_in; (void)out_size; (void)ws_size;
    const float* x        = (const float*)d_in[0];
    const int*   ei       = (const int*)  d_in[1];
    const float* in_w     = (const float*)d_in[2];
    const float* in_b     = (const float*)d_in[3];
    const float* parent_w = (const float*)d_in[4];
    const float* child_w  = (const float*)d_in[5];
    const float* self_w   = (const float*)d_in[6];
    const float* self_b   = (const float*)d_in[7];
    const float* ln_g     = (const float*)d_in[8];
    const float* ln_b     = (const float*)d_in[9];
    const float* head_w1  = (const float*)d_in[10];
    const float* head_b1  = (const float*)d_in[11];
    const float* head_w2  = (const float*)d_in[12];
    const float* head_b2  = (const float*)d_in[13];
    float* out = (float*)d_out;

    const int N = in_sizes[0] / HDIM;
    const int E = in_sizes[1] / 2;
    const int L = 3;

    // ---- workspace carve-up -------------------------------------------------
    size_t off = 0;
    char* base = (char*)d_ws;
    auto alloc = [&](size_t bytes) -> void* {
        void* p = base + off;
        off += (bytes + 255) & ~(size_t)255;
        return p;
    };
    unsigned short* bigb = (unsigned short*)alloc((size_t)N * BIGLD * 2); // bf16 [h|aggP|aggC]
    unsigned short* xb   = (unsigned short*)alloc((size_t)N * HDIM * 2);  // bf16 input x
    unsigned short* tmpb = (unsigned short*)alloc((size_t)N * HDIM * 2);  // bf16 y1
    float* hbuf = (float*)alloc((size_t)N * HDIM * 4);                    // f32 h (residual/LN)
    float* tmpf = (float*)alloc((size_t)N * HDIM * 4);                    // f32 layer pre-LN
    unsigned int* pIn  = (unsigned int*)alloc(256 * 256 * 2);
    unsigned int* pLay = (unsigned int*)alloc((size_t)L * 768 * 256 * 2);
    unsigned int* pH1  = (unsigned int*)alloc(256 * 256 * 2);
    unsigned int* pH2  = (unsigned int*)alloc(256 * 128 * 2);
    int* cntD = (int*)alloc((size_t)N * 4);
    int* cntS = (int*)alloc((size_t)N * 4);
    int* rpD  = (int*)alloc((size_t)(N + 1) * 4);
    int* rpS  = (int*)alloc((size_t)(N + 1) * 4);
    int* curD = (int*)alloc((size_t)N * 4);
    int* curS = (int*)alloc((size_t)N * 4);
    int* adjD = (int*)alloc((size_t)E * 4);
    int* adjS = (int*)alloc((size_t)E * 4);

    const int packThreads256 = 8 * 16 * 8 * 32;   // 256x256 source
    const int packThreads128 = 8 * 8 * 8 * 32;    // 256x128 source

    // ---- weight packing (bf16, lane-major WMMA B layout) --------------------
    pack_b_kernel<<<(packThreads256 + 255) / 256, 256, 0, stream>>>(
        in_w, 256, 256, 256, pIn, 0, 8);
    const size_t layStride = 24 * 16 * 8 * 32;    // uints per stacked layer
    for (int l = 0; l < L; ++l) {
        pack_b_kernel<<<(packThreads256 + 255) / 256, 256, 0, stream>>>(
            self_w + (size_t)l * 256 * 256, 256, 256, 256, pLay + l * layStride, 0, 24);
        pack_b_kernel<<<(packThreads256 + 255) / 256, 256, 0, stream>>>(
            parent_w + (size_t)l * 256 * 256, 256, 256, 256, pLay + l * layStride, 8, 24);
        pack_b_kernel<<<(packThreads256 + 255) / 256, 256, 0, stream>>>(
            child_w + (size_t)l * 256 * 256, 256, 256, 256, pLay + l * layStride, 16, 24);
    }
    pack_b_kernel<<<(packThreads256 + 255) / 256, 256, 0, stream>>>(
        head_w1, 256, 256, 256, pH1, 0, 8);
    pack_b_kernel<<<(packThreads128 + 255) / 256, 256, 0, stream>>>(
        head_w2, 128, 256, 128, pH2, 0, 8);

    // ---- bf16 copy of x (GEMM A operand) ------------------------------------
    const long n4 = (long)N * HDIM / 4;
    cast_kernel<<<(unsigned)((n4 + 255) / 256), 256, 0, stream>>>(
        (const float4*)x, (uint2*)xb, n4);

    // ---- CSR build ----------------------------------------------------------
    hipMemsetAsync(cntD, 0, (size_t)N * 4, stream);
    hipMemsetAsync(cntS, 0, (size_t)N * 4, stream);
    count_kernel<<<(E + 255) / 256, 256, 0, stream>>>(ei, E, cntD, cntS);
    scan_kernel<<<1, 1024, 0, stream>>>(cntD, rpD, N);
    scan_kernel<<<1, 1024, 0, stream>>>(cntS, rpS, N);
    copy2_kernel<<<(N + 255) / 256, 256, 0, stream>>>(rpD, curD, rpS, curS, N);
    scatter_kernel<<<(E + 255) / 256, 256, 0, stream>>>(ei, E, curD, adjD, curS, adjS);

    // ---- forward pass -------------------------------------------------------
    dim3 blk(256);
    dim3 g256(4, (unsigned)((N / 16 + 7) / 8));   // Ncols=256
    dim3 g128(2, (unsigned)((N / 16 + 7) / 8));   // Ncols=128

    // h = x @ in_w + in_b  -> hbuf (f32) + bigb[:,0:256] (bf16)
    wmma_gemm_kernel<<<g256, blk, 0, stream>>>(
        xb, 256, (const uint4*)pIn, 8, in_b, hbuf, HDIM, bigb, BIGLD, N, 2 | 4);

    for (int l = 0; l < L; ++l) {
        agg_kernel<<<(N + 7) / 8, 256, 0, stream>>>(bigb, rpD, adjD, rpS, adjS, N);
        // tmpf = gelu([h|aggP|aggC] @ [self;parent;child] + self_b)
        wmma_gemm_kernel<<<g256, blk, 0, stream>>>(
            bigb, BIGLD, (const uint4*)(pLay + l * layStride), 24,
            self_b + (size_t)l * 256, tmpf, HDIM, nullptr, 0, N, 1 | 2);
        ln_kernel<<<(N + 7) / 8, 256, 0, stream>>>(
            hbuf, tmpf, bigb, ln_g + (size_t)l * 256, ln_b + (size_t)l * 256, N);
    }

    // y1 = gelu(h @ head_w1 + b1) -> bf16 only (consumed by head2)
    wmma_gemm_kernel<<<g256, blk, 0, stream>>>(
        bigb, BIGLD, (const uint4*)pH1, 8, head_b1, nullptr, 0, tmpb, HDIM, N, 1 | 4);
    // out = y1 @ head_w2 + b2  (f32 final output)
    wmma_gemm_kernel<<<g128, blk, 0, stream>>>(
        tmpb, HDIM, (const uint4*)pH2, 8, head_b2, out, 128, nullptr, 0, N, 2);
}
